// MultiheadAttention_5669356835544
// MI455X (gfx1250) — compile-verified, matrix operand order checked by reading, not run on hardware
//
#include <hip/hip_runtime.h>
#include <hip/hip_bf16.h>

// MHA forward for MI455X (gfx1250, wave32, WMMA).
// B=2, S=2048, E=1024, H=16, D=64.
// d_out layout: [out (B*S*E f32)] ++ [att (B*H*S*S f32)]
// d_ws  layout: [qkv f16: [3][B][H][S][D]] ++ [ctx f16: [B*S][E]]

#define EDIM 1024
#define HNUM 16
#define DDIM 64
#define BDIM 2
#define SDIM 2048

typedef __attribute__((ext_vector_type(16))) _Float16 v16h;
typedef __attribute__((ext_vector_type(8)))  _Float16 v8h;
typedef __attribute__((ext_vector_type(8)))  float    v8f;
typedef __attribute__((ext_vector_type(4)))  float    v4f;

// ---------------------------------------------------------------------------
// WMMA helpers (CDNA5 16x16x32 f16 -> f32)
// ---------------------------------------------------------------------------
static __device__ __forceinline__ v8f wmma_f16(v16h a, v16h b, v8f c) {
  // 8 args: (neg_a, A, neg_b, B, c_mod, C, reuse_a, reuse_b)
  return __builtin_amdgcn_wmma_f32_16x16x32_f16(false, a, false, b,
                                                (short)0, c, false, false);
}

// Load a 16x32 f16 A/B fragment from an LDS tile where the fragment's
// "row" dimension (M for A, N for B) is the tile row and the 32 K values
// are contiguous within the row starting at k0.
// ISA layout (16-bit A 16x32): lanes 0-15 hold M=lane with K = {0..7,16..23},
// lanes 16-31 hold M=lane-16 with K = {8..15,24..31}.
static __device__ __forceinline__ v16h frag16x32(const _Float16* base,
                                                 int row0, int k0, int ld) {
  const int lane = threadIdx.x & 31;
  const int r    = row0 + (lane & 15);
  const int h8   = (lane >> 4) << 3;   // 0 or 8
  const _Float16* p = base + r * ld + k0 + h8;
  v8h lo = *(const v8h*)(p);           // K = h8 .. h8+7
  v8h hi = *(const v8h*)(p + 16);      // K = 16+h8 .. 16+h8+7
  return __builtin_shufflevector(lo, hi, 0, 1, 2, 3, 4, 5, 6, 7,
                                 8, 9, 10, 11, 12, 13, 14, 15);
}

// Async global->LDS copy of 16 bytes (CDNA5, tracked by ASYNCcnt).
// LDS generic pointers: low 32 bits are the LDS byte address (aperture rules).
static __device__ __forceinline__ void async_copy_b128(void* lds_ptr,
                                                       const void* gptr) {
  unsigned lds_addr = (unsigned)(size_t)lds_ptr;
  asm volatile("global_load_async_to_lds_b128 %0, %1, off"
               :: "v"(lds_addr), "v"(gptr)
               : "memory");
}
static __device__ __forceinline__ void wait_asynccnt0() {
  asm volatile("s_wait_asynccnt 0x0" ::: "memory");
}

// ---------------------------------------------------------------------------
// Kernel 1: fused QKV projection.  P[m, n] = X[m, :] . W[n, :] + bias[n]
// Block tile 128x64; 8 waves as 4x2; 2x2 subtiles per wave (4 WMMA/stage).
// Output written f16, head-major: qkv[proj][b][h][s][d].
// ---------------------------------------------------------------------------
__global__ __launch_bounds__(256) void qkv_proj_kernel(
    const float* __restrict__ q, const float* __restrict__ k,
    const float* __restrict__ v, const float* __restrict__ W,
    const float* __restrict__ bias, _Float16* __restrict__ qkv) {
  __shared__ __align__(16) _Float16 As[128][40];
  __shared__ __align__(16) _Float16 Bs[64][40];

  const int tid  = threadIdx.x;
  const int lane = tid & 31;
  const int wave = tid >> 5;
  const int m0   = blockIdx.y * 128;       // row in [0, B*S)
  const int n0g  = blockIdx.x * 64;        // col in [0, 3E)
  const int proj = n0g >> 10;              // 0=q,1=k,2=v (64 | 1024)
  const int n0   = n0g & (EDIM - 1);       // col within projection

  const float* X = (proj == 0) ? q : (proj == 1) ? k : v;

  const int wm = wave >> 1;                // 0..3 -> subtile rows {2wm,2wm+1}
  const int wn = wave & 1;                 // 0..1 -> subtile cols {2wn,2wn+1}
  v8f acc00 = {}, acc01 = {}, acc10 = {}, acc11 = {};

  const int ar  = tid >> 1;                // A staging row 0..127
  const int ac  = (tid & 1) * 16;          // A staging col {0,16}
  const int br  = tid >> 2;                // B staging row 0..63
  const int bc  = (tid & 3) * 8;           // B staging col {0,8,16,24}

  for (int k0 = 0; k0 < EDIM; k0 += 32) {
    {  // stage X tile 128x32, f32 -> f16
      const float* src = X + (size_t)(m0 + ar) * EDIM + k0 + ac;
      __builtin_prefetch(src + 32, 0, 0);
      v4f x0 = *(const v4f*)(src);
      v4f x1 = *(const v4f*)(src + 4);
      v4f x2 = *(const v4f*)(src + 8);
      v4f x3 = *(const v4f*)(src + 12);
      _Float16* dst = &As[ar][ac];
#pragma unroll
      for (int i = 0; i < 4; ++i) {
        dst[i]      = (_Float16)x0[i];
        dst[4 + i]  = (_Float16)x1[i];
        dst[8 + i]  = (_Float16)x2[i];
        dst[12 + i] = (_Float16)x3[i];
      }
    }
    {  // stage W tile 64x32 (rows n0g..n0g+63), f32 -> f16
      const float* src = W + (size_t)(n0g + br) * EDIM + k0 + bc;
      v4f a = *(const v4f*)(src);
      v4f b = *(const v4f*)(src + 4);
      _Float16* dst = &Bs[br][bc];
#pragma unroll
      for (int i = 0; i < 4; ++i) {
        dst[i] = (_Float16)a[i];
        dst[4 + i] = (_Float16)b[i];
      }
    }
    __syncthreads();
    v16h a0 = frag16x32(&As[0][0], (wm * 2 + 0) * 16, 0, 40);
    v16h a1 = frag16x32(&As[0][0], (wm * 2 + 1) * 16, 0, 40);
    v16h b0 = frag16x32(&Bs[0][0], (wn * 2 + 0) * 16, 0, 40);
    v16h b1 = frag16x32(&Bs[0][0], (wn * 2 + 1) * 16, 0, 40);
    acc00 = wmma_f16(a0, b0, acc00);
    acc01 = wmma_f16(a0, b1, acc01);
    acc10 = wmma_f16(a1, b0, acc10);
    acc11 = wmma_f16(a1, b1, acc11);
    __syncthreads();
  }

  // C/D layout: lane<16: (M=r, N=lane); lane>=16: (M=8+r, N=lane-16)
  const int colN  = lane & 15;
  const int rowHi = (lane >> 4) * 8;
  const v8f* accs[4] = {&acc00, &acc01, &acc10, &acc11};
#pragma unroll
  for (int st = 0; st < 4; ++st) {
    const int i = st >> 1, j = st & 1;
    const v8f& a = *accs[st];
#pragma unroll
    for (int r = 0; r < 8; ++r) {
      const int gm = m0 + (wm * 2 + i) * 16 + rowHi + r;
      const int ec = n0 + (wn * 2 + j) * 16 + colN;
      const float val = a[r] + bias[n0g + (wn * 2 + j) * 16 + colN];
      const int b_ = gm >> 11, s_ = gm & (SDIM - 1);
      const int h_ = ec >> 6, d_ = ec & 63;
      const size_t di =
          ((((size_t)proj * BDIM + b_) * HNUM + h_) * SDIM + s_) * DDIM + d_;
      qkv[di] = (_Float16)val;
    }
  }
}

// ---------------------------------------------------------------------------
// Kernel 2: fused attention.  One block = 64-query strip of one head.
// Pass 1: WMMA sweep over all K-tiles -> exact per-row (max m, sum l) via
//         online update through a small LDS score tile (no global traffic).
// Pass 2: re-sweep, recompute identical scores, write normalized
//         att = exp(x-m)/l to d_out (att is WRITE-ONLY in HBM), stash f16
//         probs in LDS and immediately accumulate ctx += P x V with WMMA.
// ---------------------------------------------------------------------------
__global__ __launch_bounds__(256) void attention_kernel(
    const _Float16* __restrict__ qkv, const unsigned char* __restrict__ mask,
    float* __restrict__ att, _Float16* __restrict__ ctx) {
  __shared__ __align__(16) _Float16 Qs[64][72];
  __shared__ __align__(16) _Float16 Ks[64][72];
  __shared__ __align__(16) _Float16 Vt[64][72];   // V transposed [d][sk]
  __shared__ __align__(16) _Float16 Pt[64][72];   // normalized probs f16
  __shared__ float St[64][68];
  __shared__ float red4[64][4];
  __shared__ float sm[64];
  __shared__ float sl[64];

  const int tid  = threadIdx.x;
  const int lane = tid & 31;
  const int wave = tid >> 5;
  const int sq0  = blockIdx.x * 64;
  const int bh   = blockIdx.y;
  const int b_   = bh >> 4, h_ = bh & (HNUM - 1);

  const _Float16* Qg =
      qkv + (((size_t)0 * BDIM + b_) * HNUM + h_) * (size_t)SDIM * DDIM;
  const _Float16* Kg =
      qkv + (((size_t)1 * BDIM + b_) * HNUM + h_) * (size_t)SDIM * DDIM;
  const _Float16* Vg =
      qkv + (((size_t)2 * BDIM + b_) * HNUM + h_) * (size_t)SDIM * DDIM;
  const unsigned char* mB = mask + (size_t)b_ * SDIM * SDIM;
  float* attBH = att + (size_t)bh * SDIM * SDIM;

  const int srow = tid >> 2;           // staging / reduction row 0..63
  const int sc16 = (tid & 3) * 16;     // staging col {0,16,32,48}
  const int seg  = tid & 3;            // 16-col segment per row

  {  // async-stage Q strip once
    async_copy_b128(&Qs[srow][sc16],
                    Qg + (size_t)(sq0 + srow) * DDIM + sc16);
    async_copy_b128(&Qs[srow][sc16 + 8],
                    Qg + (size_t)(sq0 + srow) * DDIM + sc16 + 8);
    wait_asynccnt0();
  }
  if (tid < 64) { sm[tid] = -3.0e38f; sl[tid] = 0.0f; }
  __syncthreads();

  const int ms0 = wave >> 2, ns0 = wave & 3;
  const int ms1 = ms0 + 2;
  const int colN  = lane & 15;
  const int rowHi = (lane >> 4) * 8;
  const float scale = 0.125f;   // D^-0.5

  // ------------------------- pass 1: row stats -------------------------
  for (int sk0 = 0; sk0 < SDIM; sk0 += 64) {
    {  // async-stage K tile
      async_copy_b128(&Ks[srow][sc16],
                      Kg + (size_t)(sk0 + srow) * DDIM + sc16);
      async_copy_b128(&Ks[srow][sc16 + 8],
                      Kg + (size_t)(sk0 + srow) * DDIM + sc16 + 8);
      wait_asynccnt0();
    }
    __syncthreads();

    v8f acc0 = {}, acc1 = {};
#pragma unroll
    for (int kk = 0; kk < DDIM; kk += 32) {
      v16h bb = frag16x32(&Ks[0][0], ns0 * 16, kk, 72);
      v16h a0 = frag16x32(&Qs[0][0], ms0 * 16, kk, 72);
      acc0 = wmma_f16(a0, bb, acc0);
      v16h a1 = frag16x32(&Qs[0][0], ms1 * 16, kk, 72);
      acc1 = wmma_f16(a1, bb, acc1);
    }
#pragma unroll
    for (int r = 0; r < 8; ++r) {
      {
        const int lr = ms0 * 16 + rowHi + r;
        const int lc = ns0 * 16 + colN;
        float val = acc0[r] * scale;
        if (mB[(size_t)(sq0 + lr) * SDIM + sk0 + lc]) val = -1.0e9f;
        St[lr][lc] = val;
      }
      {
        const int lr = ms1 * 16 + rowHi + r;
        const int lc = ns0 * 16 + colN;
        float val = acc1[r] * scale;
        if (mB[(size_t)(sq0 + lr) * SDIM + sk0 + lc]) val = -1.0e9f;
        St[lr][lc] = val;
      }
    }
    __syncthreads();

    float pmax = -3.0e38f;
#pragma unroll
    for (int i = 0; i < 16; ++i) pmax = fmaxf(pmax, St[srow][seg * 16 + i]);
    red4[srow][seg] = pmax;
    __syncthreads();
    if (seg == 0) {
      const float tm = fmaxf(fmaxf(red4[srow][0], red4[srow][1]),
                             fmaxf(red4[srow][2], red4[srow][3]));
      const float mo = sm[srow];
      const float mn = fmaxf(mo, tm);
      sl[srow] *= __expf(mo - mn);   // rescale running sum
      sm[srow] = mn;
    }
    __syncthreads();
    const float mrow = sm[srow];
    float psum = 0.0f;
#pragma unroll
    for (int i = 0; i < 16; ++i) psum += __expf(St[srow][seg * 16 + i] - mrow);
    red4[srow][seg] = psum;
    __syncthreads();
    if (seg == 0)
      sl[srow] += red4[srow][0] + red4[srow][1] + red4[srow][2] + red4[srow][3];
    __syncthreads();
  }

  // preload per-lane (m, 1/l) for this wave's two subtile row groups
  float m0r[8], il0r[8], m1r[8], il1r[8];
#pragma unroll
  for (int r = 0; r < 8; ++r) {
    const int lr0 = ms0 * 16 + rowHi + r;
    const int lr1 = ms1 * 16 + rowHi + r;
    m0r[r]  = sm[lr0];
    il0r[r] = 1.0f / sl[lr0];
    m1r[r]  = sm[lr1];
    il1r[r] = 1.0f / sl[lr1];
  }
  __syncthreads();

  // --------------- pass 2: normalized att write + AV accumulate ---------------
  v8f cacc0 = {}, cacc1 = {};   // ctx subtiles (ms0,ns0), (ms1,ns0)
  for (int sk0 = 0; sk0 < SDIM; sk0 += 64) {
    {  // async-stage K tile
      async_copy_b128(&Ks[srow][sc16],
                      Kg + (size_t)(sk0 + srow) * DDIM + sc16);
      async_copy_b128(&Ks[srow][sc16 + 8],
                      Kg + (size_t)(sk0 + srow) * DDIM + sc16 + 8);
    }
    {  // stage V tile 64x64 transposed into Vt[d][sk]
      v8h v0 = *(const v8h*)(Vg + (size_t)(sk0 + srow) * DDIM + sc16);
      v8h v1 = *(const v8h*)(Vg + (size_t)(sk0 + srow) * DDIM + sc16 + 8);
#pragma unroll
      for (int i = 0; i < 8; ++i) {
        Vt[sc16 + i][srow]     = v0[i];
        Vt[sc16 + 8 + i][srow] = v1[i];
      }
    }
    wait_asynccnt0();
    __syncthreads();

    // recompute scores (bitwise-identical to pass 1)
    v8f acc0 = {}, acc1 = {};
#pragma unroll
    for (int kk = 0; kk < DDIM; kk += 32) {
      v16h bb = frag16x32(&Ks[0][0], ns0 * 16, kk, 72);
      v16h a0 = frag16x32(&Qs[0][0], ms0 * 16, kk, 72);
      acc0 = wmma_f16(a0, bb, acc0);
      v16h a1 = frag16x32(&Qs[0][0], ms1 * 16, kk, 72);
      acc1 = wmma_f16(a1, bb, acc1);
    }
    // normalize, write att (global, write-only) and Pt (LDS, f16)
#pragma unroll
    for (int r = 0; r < 8; ++r) {
      {
        const int lr = ms0 * 16 + rowHi + r;
        const int lc = ns0 * 16 + colN;
        float val = acc0[r] * scale;
        if (mB[(size_t)(sq0 + lr) * SDIM + sk0 + lc]) val = -1.0e9f;
        const float p = __expf(val - m0r[r]) * il0r[r];
        attBH[(size_t)(sq0 + lr) * SDIM + sk0 + lc] = p;
        Pt[lr][lc] = (_Float16)p;
      }
      {
        const int lr = ms1 * 16 + rowHi + r;
        const int lc = ns0 * 16 + colN;
        float val = acc1[r] * scale;
        if (mB[(size_t)(sq0 + lr) * SDIM + sk0 + lc]) val = -1.0e9f;
        const float p = __expf(val - m1r[r]) * il1r[r];
        attBH[(size_t)(sq0 + lr) * SDIM + sk0 + lc] = p;
        Pt[lr][lc] = (_Float16)p;
      }
    }
    __syncthreads();

    // ctx += P (64x64) x V (64x64)
#pragma unroll
    for (int kk = 0; kk < 64; kk += 32) {
      v16h bb = frag16x32(&Vt[0][0], ns0 * 16, kk, 72);
      v16h a0 = frag16x32(&Pt[0][0], ms0 * 16, kk, 72);
      cacc0 = wmma_f16(a0, bb, cacc0);
      v16h a1 = frag16x32(&Pt[0][0], ms1 * 16, kk, 72);
      cacc1 = wmma_f16(a1, bb, cacc1);
    }
    __syncthreads();
  }

  // write ctx f16 in [B*S][E] layout (heads interleaved)
#pragma unroll
  for (int r = 0; r < 8; ++r) {
    {
      const int sq = sq0 + ms0 * 16 + rowHi + r;
      const int d  = ns0 * 16 + colN;
      ctx[((size_t)(b_ * SDIM + sq)) * EDIM + h_ * DDIM + d] =
          (_Float16)cacc0[r];
    }
    {
      const int sq = sq0 + ms1 * 16 + rowHi + r;
      const int d  = ns0 * 16 + colN;
      ctx[((size_t)(b_ * SDIM + sq)) * EDIM + h_ * DDIM + d] =
          (_Float16)cacc1[r];
    }
  }
}

// ---------------------------------------------------------------------------
// Kernel 3: output projection.  out = ctx @ Wo^T + bo, f32 to d_out.
// M = B*S = 4096, N = E = 1024, K = E = 1024.  Block tile 128x64, 2x2/wave.
// ---------------------------------------------------------------------------
__global__ __launch_bounds__(256) void out_proj_kernel(
    const _Float16* __restrict__ ctx, const float* __restrict__ W,
    const float* __restrict__ bias, float* __restrict__ out) {
  __shared__ __align__(16) _Float16 As[128][40];
  __shared__ __align__(16) _Float16 Bs[64][40];

  const int tid  = threadIdx.x;
  const int lane = tid & 31;
  const int wave = tid >> 5;
  const int n0   = blockIdx.x * 64;
  const int m0   = blockIdx.y * 128;

  const int wm = wave >> 1;
  const int wn = wave & 1;
  v8f acc00 = {}, acc01 = {}, acc10 = {}, acc11 = {};

  const int ar = tid >> 1;            // 0..127
  const int ac = (tid & 1) * 16;      // 0,16
  const int br = tid >> 2;            // 0..63
  const int bc = (tid & 3) * 8;       // 0,8,16,24

  for (int k0 = 0; k0 < EDIM; k0 += 32) {
    // stage ctx (already f16): 128x32 via async global->LDS
    async_copy_b128(&As[ar][ac],
                    ctx + (size_t)(m0 + ar) * EDIM + k0 + ac);
    async_copy_b128(&As[ar][ac + 8],
                    ctx + (size_t)(m0 + ar) * EDIM + k0 + ac + 8);
    {  // stage Wo rows 3E+n0.., f32 -> f16
      const float* src = W + (size_t)(3 * EDIM + n0 + br) * EDIM + k0 + bc;
      v4f a = *(const v4f*)(src);
      v4f b = *(const v4f*)(src + 4);
      _Float16* dst = &Bs[br][bc];
#pragma unroll
      for (int i = 0; i < 4; ++i) {
        dst[i] = (_Float16)a[i];
        dst[4 + i] = (_Float16)b[i];
      }
    }
    wait_asynccnt0();
    __syncthreads();
    v16h a0 = frag16x32(&As[0][0], (wm * 2 + 0) * 16, 0, 40);
    v16h a1 = frag16x32(&As[0][0], (wm * 2 + 1) * 16, 0, 40);
    v16h b0 = frag16x32(&Bs[0][0], (wn * 2 + 0) * 16, 0, 40);
    v16h b1 = frag16x32(&Bs[0][0], (wn * 2 + 1) * 16, 0, 40);
    acc00 = wmma_f16(a0, b0, acc00);
    acc01 = wmma_f16(a0, b1, acc01);
    acc10 = wmma_f16(a1, b0, acc10);
    acc11 = wmma_f16(a1, b1, acc11);
    __syncthreads();
  }

  const int colN  = lane & 15;
  const int rowHi = (lane >> 4) * 8;
  const v8f* accs[4] = {&acc00, &acc01, &acc10, &acc11};
#pragma unroll
  for (int st = 0; st < 4; ++st) {
    const int i = st >> 1, j = st & 1;
    const v8f& a = *accs[st];
#pragma unroll
    for (int r = 0; r < 8; ++r) {
      const int gm = m0 + (wm * 2 + i) * 16 + rowHi + r;
      const int gn = n0 + (wn * 2 + j) * 16 + colN;
      out[(size_t)gm * EDIM + gn] = a[r] + bias[3 * EDIM + gn];
    }
  }
}

// ---------------------------------------------------------------------------
// Launcher
// ---------------------------------------------------------------------------
extern "C" void kernel_launch(void* const* d_in, const int* in_sizes, int n_in,
                              void* d_out, int out_size, void* d_ws,
                              size_t ws_size, hipStream_t stream) {
  (void)in_sizes; (void)n_in; (void)out_size; (void)ws_size;

  const float* q = (const float*)d_in[0];
  const float* k = (const float*)d_in[1];
  const float* v = (const float*)d_in[2];
  const unsigned char* mask = (const unsigned char*)d_in[3];  // bool mask
  const float* W = (const float*)d_in[4];                     // [4E][E]
  const float* bias = (const float*)d_in[5];                  // [4E]

  float* out = (float*)d_out;                                // [B*S][E]
  float* att = out + (size_t)BDIM * SDIM * EDIM;             // [B*H][S][S]

  _Float16* qkv = (_Float16*)d_ws;                           // [3][B][H][S][D]
  _Float16* ctx = qkv + (size_t)3 * BDIM * SDIM * EDIM;      // [B*S][E]

  dim3 blk(256);
  qkv_proj_kernel<<<dim3(48, 32), blk, 0, stream>>>(q, k, v, W, bias, qkv);
  attention_kernel<<<dim3(32, 32), blk, 0, stream>>>(qkv, mask, att, ctx);
  out_proj_kernel<<<dim3(16, 32), blk, 0, stream>>>(ctx, W, bias, out);
}
